// Decoder_13005160973216
// MI455X (gfx1250) — matrix operand order repaired and checked by reading, not compiled, not run on hardware
//
#include <hip/hip_runtime.h>
#include <math.h>

#define BB 64
#define SS 128
#define TT 64
#define HH 512
#define EE 256
#define VV 10000
#define XK 1792   // E + 2H + H (gate input) == E + H + 2H (pre input)

typedef __attribute__((ext_vector_type(16))) __bf16 v16bf;
typedef __attribute__((ext_vector_type(8)))  float  v8f;

struct U32x8 { uint4 lo; uint4 hi; };

static __device__ __forceinline__ unsigned short f2bf(float f) {
    unsigned int u = __builtin_bit_cast(unsigned int, f);
    unsigned int r = 0x7FFFu + ((u >> 16) & 1u);
    return (unsigned short)((u + r) >> 16);
}

static __device__ __forceinline__ v16bf load_frag(const unsigned short* p0,
                                                  const unsigned short* p1) {
    U32x8 u;
    u.lo = *(const uint4*)p0;
    u.hi = *(const uint4*)p1;
    return __builtin_bit_cast(v16bf, u);
}

// C[m,n] = sum_k A[m,k] * Bt[n,k]; A:[M,K] bf16 row-major, Bt:[N,K] bf16 row-major.
// M multiple of 32, K multiple of 32. Each wave: 32x64 output block (2 M-tiles x
// 4 N-tiles, 8 f32 accumulators) -> A frag reused 4x, B frag reused 2x.
// Bt must be readable up to ceil(N/64)*64 rows (pad allocation for ragged N).
__global__ void wmma_gemm_bf16(const unsigned short* __restrict__ A,
                               const unsigned short* __restrict__ Bt,
                               float* __restrict__ Cf,
                               unsigned short* __restrict__ Cbf,
                               int M, int N, int K, int storeBf) {
    int lane = threadIdx.x & 31;
    int wave = threadIdx.x >> 5;
    int half = lane >> 4;       // 0 or 1
    int l16  = lane & 15;
    int m0  = blockIdx.y * 32;
    int nt0 = (blockIdx.x * 8 + wave) * 4;
    if (nt0 * 16 >= N) return;  // wave-uniform; surviving waves keep EXEC all-1s

    const unsigned short* Arow0 = A + (size_t)(m0 + l16) * K;
    const unsigned short* Arow1 = A + (size_t)(m0 + 16 + l16) * K;
    const unsigned short* Brow0 = Bt + (size_t)((nt0 + 0) * 16 + l16) * K;
    const unsigned short* Brow1 = Bt + (size_t)((nt0 + 1) * 16 + l16) * K;
    const unsigned short* Brow2 = Bt + (size_t)((nt0 + 2) * 16 + l16) * K;
    const unsigned short* Brow3 = Bt + (size_t)((nt0 + 3) * 16 + l16) * K;

    const v8f z = {0.f, 0.f, 0.f, 0.f, 0.f, 0.f, 0.f, 0.f};
    v8f accA[4] = {z, z, z, z};   // rows m0..m0+15
    v8f accB[4] = {z, z, z, z};   // rows m0+16..m0+31

    for (int k0 = 0; k0 < K; k0 += 32) {
        // A 16x32 frag: elems 0..15 -> k0+16*half+i (row = l16)
        v16bf a0 = load_frag(Arow0 + k0 + 16 * half, Arow0 + k0 + 16 * half + 8);
        v16bf a1 = load_frag(Arow1 + k0 + 16 * half, Arow1 + k0 + 16 * half + 8);
        // B 32x16 frag: elems 0..15 -> k0+16*half+i (col = l16)
        v16bf b0 = load_frag(Brow0 + k0 + 16 * half, Brow0 + k0 + 16 * half + 8);
        v16bf b1 = load_frag(Brow1 + k0 + 16 * half, Brow1 + k0 + 16 * half + 8);
        v16bf b2 = load_frag(Brow2 + k0 + 16 * half, Brow2 + k0 + 16 * half + 8);
        v16bf b3 = load_frag(Brow3 + k0 + 16 * half, Brow3 + k0 + 16 * half + 8);
        accA[0] = __builtin_amdgcn_wmma_f32_16x16x32_bf16(false, a0, false, b0, (short)0, accA[0], false, false);
        accB[0] = __builtin_amdgcn_wmma_f32_16x16x32_bf16(false, a1, false, b0, (short)0, accB[0], false, false);
        accA[1] = __builtin_amdgcn_wmma_f32_16x16x32_bf16(false, a0, false, b1, (short)0, accA[1], false, false);
        accB[1] = __builtin_amdgcn_wmma_f32_16x16x32_bf16(false, a1, false, b1, (short)0, accB[1], false, false);
        accA[2] = __builtin_amdgcn_wmma_f32_16x16x32_bf16(false, a0, false, b2, (short)0, accA[2], false, false);
        accB[2] = __builtin_amdgcn_wmma_f32_16x16x32_bf16(false, a1, false, b2, (short)0, accB[2], false, false);
        accA[3] = __builtin_amdgcn_wmma_f32_16x16x32_bf16(false, a0, false, b3, (short)0, accA[3], false, false);
        accB[3] = __builtin_amdgcn_wmma_f32_16x16x32_bf16(false, a1, false, b3, (short)0, accB[3], false, false);
    }

    // D layout: VGPR r holds row (tileRow + r + 8*half), col (n0 + l16)
    #pragma unroll
    for (int j = 0; j < 4; ++j) {
        int n0 = (nt0 + j) * 16;
        if (n0 >= N) break;     // wave-uniform
        #pragma unroll
        for (int r = 0; r < 8; ++r) {
            size_t oA = (size_t)(m0 + r + 8 * half) * N + n0 + l16;
            size_t oB = (size_t)(m0 + 16 + r + 8 * half) * N + n0 + l16;
            if (storeBf) {
                Cbf[oA] = f2bf(accA[j][r]);
                Cbf[oB] = f2bf(accB[j][r]);
            } else {
                Cf[oA] = accA[j][r];
                Cf[oB] = accB[j][r];
            }
        }
    }
}

// out[c*R + r] = bf16(in[r*C + c])   (transpose + convert)
__global__ void k_transpose_bf16(const float* __restrict__ in,
                                 unsigned short* __restrict__ out,
                                 int R, int C, int n) {
    int i = blockIdx.x * blockDim.x + threadIdx.x;
    if (i >= n) return;
    int r = i / C, c = i % C;
    out[(size_t)c * R + r] = f2bf(in[i]);
}

// Wg_t[j, 0:1280] = W_ih[j,:], Wg_t[j, 1280:1792] = W_hh[j,:]
__global__ void k_build_wg(const float* __restrict__ Wih,
                           const float* __restrict__ Whh,
                           unsigned short* __restrict__ out) {
    int i = blockIdx.x * blockDim.x + threadIdx.x;
    if (i >= 4 * HH * XK) return;
    int j = i / XK, k = i % XK;
    float v = (k < EE + 2 * HH) ? Wih[(size_t)j * (EE + 2 * HH) + k]
                                : Whh[(size_t)j * HH + (k - EE - 2 * HH)];
    out[i] = f2bf(v);
}

__global__ void k_f32_to_bf16(const float* __restrict__ in,
                              unsigned short* __restrict__ out, int n) {
    int i = blockIdx.x * blockDim.x + threadIdx.x;
    if (i < n) out[i] = f2bf(in[i]);
}

// embs_bf[(b*T+t)*E + e] = bf16(table[trg[b*T+t]*E + e])
__global__ void k_gather_embs(const int* __restrict__ trg,
                              const float* __restrict__ table,
                              unsigned short* __restrict__ out) {
    int i = blockIdx.x * blockDim.x + threadIdx.x;
    if (i >= BB * TT * EE) return;
    int bt = i / EE, e = i % EE;
    int id = trg[bt];
    out[i] = f2bf(table[(size_t)id * EE + e]);
}

__global__ void k_init_hc(const float* __restrict__ h0, const float* __restrict__ c0,
                          float* __restrict__ hf, unsigned short* __restrict__ hbf,
                          float* __restrict__ cf) {
    int i = blockIdx.x * blockDim.x + threadIdx.x;
    if (i >= BB * HH) return;
    float h = h0[i];
    hf[i] = h; hbf[i] = f2bf(h); cf[i] = c0[i];
}

// Per-step attention: energies, masked softmax, context; builds xcat/precat (bf16).
// Uses CDNA5 async global->LDS copies (ASYNCcnt) to stage the q row and energy
// weights into LDS without touching VGPRs.
__global__ void k_attn(const float* __restrict__ q,          // [B,H]
                       const float* __restrict__ pk,         // [B,S,H]
                       const float* __restrict__ enc,        // [B,S,2H]
                       const float* __restrict__ we,         // [H]
                       const int* __restrict__ slen,         // [B]
                       const unsigned short* __restrict__ embs, // [B*T,E]
                       const unsigned short* __restrict__ hbf,  // [B,H]
                       float* __restrict__ ctx,              // [B,2H]
                       unsigned short* __restrict__ xcat,    // [B,XK] = [emb,ctx,h]
                       unsigned short* __restrict__ precat,  // [B,XK] = [emb,h_new,ctx]
                       int t) {
    __shared__ float sq[HH], swe[HH], sa[SS], sred[SS];
    int b = blockIdx.x, tid = threadIdx.x;                   // 128 threads

    // async global->LDS: 128 lanes x 16B = 512 floats per buffer
    {
        const float* gq = q + (size_t)b * HH + tid * 4;
        const float* gw = we + tid * 4;
        unsigned oq = (unsigned)(size_t)&sq[tid * 4];   // low 32 bits = LDS offset
        unsigned ow = (unsigned)(size_t)&swe[tid * 4];
        asm volatile("global_load_async_to_lds_b128 %0, %1, off\n\t"
                     "global_load_async_to_lds_b128 %2, %3, off"
                     :: "v"(oq), "v"(gq), "v"(ow), "v"(gw)
                     : "memory");
        asm volatile("s_wait_asynccnt 0" ::: "memory");
    }
    __syncthreads();

    const float* pkb = pk + ((size_t)b * SS + tid) * HH;
    float e = 0.f;
    for (int h = 0; h < HH; ++h) e += swe[h] * tanhf(sq[h] + pkb[h]);
    int len = slen[b];
    e = (tid < len) ? e : -1e9f;

    sred[tid] = e; __syncthreads();
    for (int o = 64; o > 0; o >>= 1) { if (tid < o) sred[tid] = fmaxf(sred[tid], sred[tid + o]); __syncthreads(); }
    float mx = sred[0]; __syncthreads();
    float pexp = __expf(e - mx);
    sred[tid] = pexp; __syncthreads();
    for (int o = 64; o > 0; o >>= 1) { if (tid < o) sred[tid] += sred[tid + o]; __syncthreads(); }
    sa[tid] = pexp / sred[0];
    __syncthreads();

    const float* encb = enc + (size_t)b * SS * 2 * HH;
    for (int d = tid; d < 2 * HH; d += SS) {
        float acc = 0.f;
        for (int s = 0; s < SS; ++s) acc += sa[s] * encb[(size_t)s * 2 * HH + d];
        ctx[b * 2 * HH + d] = acc;
        unsigned short bv = f2bf(acc);
        xcat  [(size_t)b * XK + EE + d]      = bv;   // x = [emb, ctx, h]
        precat[(size_t)b * XK + EE + HH + d] = bv;   // pre = [emb, h_new, ctx]
    }
    const unsigned short* eb = embs + ((size_t)b * TT + t) * EE;
    for (int i = tid; i < EE; i += SS) {
        unsigned short v = eb[i];
        xcat[(size_t)b * XK + i] = v;
        precat[(size_t)b * XK + i] = v;
    }
    for (int i = tid; i < HH; i += SS)
        xcat[(size_t)b * XK + EE + 2 * HH + i] = hbf[b * HH + i];
}

__global__ void k_lstm(const float* __restrict__ g,          // [B,4H]
                       const float* __restrict__ bih, const float* __restrict__ bhh,
                       float* __restrict__ hf, unsigned short* __restrict__ hbf,
                       float* __restrict__ cf, unsigned short* __restrict__ precat) {
    int i = blockIdx.x * blockDim.x + threadIdx.x;
    if (i >= BB * HH) return;
    int b = i / HH, j = i % HH;
    const float* gb = g + (size_t)b * 4 * HH;
    float ig = gb[j]            + bih[j]            + bhh[j];
    float fg = gb[HH + j]       + bih[HH + j]       + bhh[HH + j];
    float gg = gb[2 * HH + j]   + bih[2 * HH + j]   + bhh[2 * HH + j];
    float og = gb[3 * HH + j]   + bih[3 * HH + j]   + bhh[3 * HH + j];
    ig = 1.f / (1.f + __expf(-ig));
    fg = 1.f / (1.f + __expf(-fg));
    og = 1.f / (1.f + __expf(-og));
    gg = tanhf(gg);
    float cn = fg * cf[i] + ig * gg;
    float hn = og * tanhf(cn);
    cf[i] = cn; hf[i] = hn;
    unsigned short hb = f2bf(hn);
    hbf[i] = hb;
    precat[(size_t)b * XK + EE + j] = hb;
}

__global__ void k_logsoftmax(const float* __restrict__ logits, // [B,V]
                             float* __restrict__ out, int t) {
    __shared__ float sred[256];
    int b = blockIdx.x, tid = threadIdx.x;                    // 256 threads
    const float* lb = logits + (size_t)b * VV;
    float mx = -1e30f;
    for (int v = tid; v < VV; v += 256) mx = fmaxf(mx, lb[v]);
    sred[tid] = mx; __syncthreads();
    for (int o = 128; o > 0; o >>= 1) { if (tid < o) sred[tid] = fmaxf(sred[tid], sred[tid + o]); __syncthreads(); }
    mx = sred[0]; __syncthreads();
    float s = 0.f;
    for (int v = tid; v < VV; v += 256) s += __expf(lb[v] - mx);
    sred[tid] = s; __syncthreads();
    for (int o = 128; o > 0; o >>= 1) { if (tid < o) sred[tid] += sred[tid + o]; __syncthreads(); }
    float lz = mx + __logf(sred[0]);
    float* ob = out + (size_t)b * VV * TT;
    for (int v = tid; v < VV; v += 256) ob[(size_t)v * TT + t] = lb[v] - lz;
}

__global__ void k_tail(const float* __restrict__ hf, const float* __restrict__ cf,
                       float* __restrict__ out) {
    int i = blockIdx.x * blockDim.x + threadIdx.x;
    if (i >= BB * HH) return;
    size_t base = (size_t)BB * VV * TT;
    out[base + i] = hf[i];
    out[base + BB * HH + i] = cf[i];
}

static inline void gemm(const unsigned short* A, const unsigned short* Bt,
                        float* Cf, unsigned short* Cbf,
                        int M, int N, int K, int storeBf, hipStream_t s) {
    dim3 grid((N + 511) / 512, M / 32);   // wave = 32x64 block, 8 waves/block
    wmma_gemm_bf16<<<grid, 256, 0, s>>>(A, Bt, Cf, Cbf, M, N, K, storeBf);
}

extern "C" void kernel_launch(void* const* d_in, const int* in_sizes, int n_in,
                              void* d_out, int out_size, void* d_ws, size_t ws_size,
                              hipStream_t stream) {
    const int*   trg    = (const int*)d_in[0];
    const float* enc    = (const float*)d_in[1];
    const int*   slen   = (const int*)d_in[2];
    const float* h0     = (const float*)d_in[3];
    const float* c0     = (const float*)d_in[4];
    const float* table  = (const float*)d_in[5];
    const float* Wkey   = (const float*)d_in[6];
    const float* Wquery = (const float*)d_in[7];
    const float* we     = (const float*)d_in[8];
    const float* W_ih   = (const float*)d_in[9];
    const float* W_hh   = (const float*)d_in[10];
    const float* b_ih   = (const float*)d_in[11];
    const float* b_hh   = (const float*)d_in[12];
    const float* Wpre   = (const float*)d_in[13];
    const float* Wgen   = (const float*)d_in[14];
    float* out = (float*)d_out;

    const int VPAD = (VV + 63) & ~63;   // 10048: pad Wgen_t rows for ragged N

    char* p = (char*)d_ws;
    auto carve = [&](size_t bytes) -> char* {
        char* r = p; p += (bytes + 255) & ~(size_t)255; return r;
    };
    unsigned short* Wq_t   = (unsigned short*)carve((size_t)HH * HH * 2);          // [512,512]
    unsigned short* Wk_t   = (unsigned short*)carve((size_t)HH * 2 * HH * 2);      // [512,1024]
    unsigned short* Wpre_t = (unsigned short*)carve((size_t)HH * XK * 2);          // [512,1792]
    unsigned short* Wgen_t = (unsigned short*)carve((size_t)VPAD * HH * 2);        // [10048,512]
    unsigned short* Wg_t   = (unsigned short*)carve((size_t)4 * HH * XK * 2);      // [2048,1792]
    unsigned short* encbf  = (unsigned short*)carve((size_t)BB * SS * 2 * HH * 2);
    float*          pkey   = (float*)carve((size_t)BB * SS * HH * 4);
    unsigned short* embs   = (unsigned short*)carve((size_t)BB * TT * EE * 2);
    float*          hf     = (float*)carve((size_t)BB * HH * 4);
    unsigned short* hbf    = (unsigned short*)carve((size_t)BB * HH * 2);
    float*          cf     = (float*)carve((size_t)BB * HH * 4);
    float*          qbuf   = (float*)carve((size_t)BB * HH * 4);
    float*          ctx    = (float*)carve((size_t)BB * 2 * HH * 4);
    unsigned short* xcat   = (unsigned short*)carve((size_t)BB * XK * 2);
    unsigned short* precat = (unsigned short*)carve((size_t)BB * XK * 2);
    float*          gbuf   = (float*)carve((size_t)BB * 4 * HH * 4);
    unsigned short* prebf  = (unsigned short*)carve((size_t)BB * HH * 2);
    float*          logits = (float*)carve((size_t)BB * VV * 4);

    // ---- one-time prep (re-run every call; deterministic) ----
    {
        int n;
        n = HH * HH;          k_transpose_bf16<<<(n + 255) / 256, 256, 0, stream>>>(Wquery, Wq_t, HH, HH, n);
        n = 2 * HH * HH;      k_transpose_bf16<<<(n + 255) / 256, 256, 0, stream>>>(Wkey, Wk_t, 2 * HH, HH, n);
        n = XK * HH;          k_transpose_bf16<<<(n + 255) / 256, 256, 0, stream>>>(Wpre, Wpre_t, XK, HH, n);
        n = HH * VV;          k_transpose_bf16<<<(n + 255) / 256, 256, 0, stream>>>(Wgen, Wgen_t, HH, VV, n);
        n = 4 * HH * XK;      k_build_wg<<<(n + 255) / 256, 256, 0, stream>>>(W_ih, W_hh, Wg_t);
        n = BB * SS * 2 * HH; k_f32_to_bf16<<<(n + 255) / 256, 256, 0, stream>>>(enc, encbf, n);
        n = BB * TT * EE;     k_gather_embs<<<(n + 255) / 256, 256, 0, stream>>>(trg, table, embs);
        n = BB * HH;          k_init_hc<<<(n + 255) / 256, 256, 0, stream>>>(h0, c0, hf, hbf, cf);
        // proj_key = encoder_hidden @ Wkey : [8192,1024] x [1024->512]
        gemm(encbf, Wk_t, pkey, nullptr, BB * SS, HH, 2 * HH, 0, stream);
    }

    // ---- teacher-forced scan ----
    for (int t = 0; t < TT; ++t) {
        // q = h @ Wquery
        gemm(hbf, Wq_t, qbuf, nullptr, BB, HH, HH, 0, stream);
        // attention + build concat inputs
        k_attn<<<BB, 128, 0, stream>>>(qbuf, pkey, enc, we, slen, embs, hbf,
                                       ctx, xcat, precat, t);
        // gates = [emb,ctx,h] @ [W_ih|W_hh]^T
        gemm(xcat, Wg_t, gbuf, nullptr, BB, 4 * HH, XK, 0, stream);
        // LSTM pointwise; fills precat h-slot
        k_lstm<<<(BB * HH + 255) / 256, 256, 0, stream>>>(gbuf, b_ih, b_hh,
                                                          hf, hbf, cf, precat);
        // pre = [emb,h_new,ctx] @ Wpre  (bf16 out)
        gemm(precat, Wpre_t, nullptr, prebf, BB, HH, XK, 1, stream);
        // logits = pre @ Wgen
        gemm(prebf, Wgen_t, logits, nullptr, BB, VV, HH, 0, stream);
        // log_softmax -> out[b, :, t]
        k_logsoftmax<<<BB, 256, 0, stream>>>(logits, out, t);
    }

    // final h, c appended after [B,V,T] block
    k_tail<<<(BB * HH + 255) / 256, 256, 0, stream>>>(hf, cf, out);
}
